// RNNModel_2018634629151
// MI455X (gfx1250) — compile-verified, matrix operand order checked by reading, not run on hardware
//
#include <hip/hip_runtime.h>
#include <math.h>

// ---------------------------------------------------------------------------
// Problem constants (from reference): T=256, B=25, V=32000, H=1024, C=2
// Phase 1:  P[m,n] = X[m,:] @ W_xh[:,n]   (M=6400, N=1024, K=32000)  -- bf16 WMMA
//           B-tiles staged to LDS via the Tensor Data Mover (TENSORcnt)
// Phase 2:  h = tanh(P_t + h @ W_hh + b)  256 serial steps           -- bf16 WMMA
// Phase 3:  softmax(mean_b(h_T) @ fc_w + fc_b)
// ---------------------------------------------------------------------------

typedef __bf16 bf16;
typedef __attribute__((ext_vector_type(8)))  __bf16 v8bf;
typedef __attribute__((ext_vector_type(16))) __bf16 v16bf;
typedef __attribute__((ext_vector_type(8)))  float  v8f;
typedef __attribute__((ext_vector_type(4)))  unsigned v4u;
typedef __attribute__((ext_vector_type(8)))  int      v8i;
typedef __attribute__((ext_vector_type(4)))  int      v4i;

#define DEVINL __device__ __forceinline__

#if __has_builtin(__builtin_amdgcn_tensor_load_to_lds)
#define HAVE_TDM 1
#else
#define HAVE_TDM 0
#endif

static constexpr int TT  = 256;
static constexpr int BSZ = 25;
static constexpr int V   = 32000;
static constexpr int H   = 1024;
static constexpr int M   = TT * BSZ;   // 6400

// A/B fragment: lane<16 -> row(lane), K {0..7,16..23}; lane>=16 -> K {8..15,24..31}
DEVINL v16bf make_frag(const bf16* p) {
  v8bf lo = *(const v8bf*)(p);
  v8bf hi = *(const v8bf*)(p + 16);
  return __builtin_shufflevector(lo, hi, 0,1,2,3,4,5,6,7,8,9,10,11,12,13,14,15);
}

DEVINL v8f wmma_bf16(v16bf a, v16bf b, v8f c) {
  // (neg_a, A, neg_b, B, c_mod, C, reuse_a, reuse_b)
  return __builtin_amdgcn_wmma_f32_16x16x32_bf16(false, a, false, b, (short)0, c,
                                                 false, false);
}

// TDM: DMA one 128x32 bf16 tile of Wt (row stride V) into LDS with 16B row pad
// (64B data + 16B pad = 80B stride == LDA*2). Issued once per wave, TENSORcnt.
DEVINL void tdm_load_wt_tile(const bf16* gsrc, bf16* ldst) {
#if HAVE_TDM
  unsigned long long ga = (unsigned long long)(uintptr_t)gsrc;
  unsigned lds = (unsigned)(uintptr_t)ldst;  // low 32 bits = LDS byte offset
  v4u g0 = {1u,                              // count=1, user-mode, no gather
            lds,                             // lds_addr
            (unsigned)ga,                    // global_addr[31:0]
            (unsigned)(ga >> 32) | (2u << 30)};  // global_addr[56:32] | type=2
  const unsigned p0 = (1u << 16)   // data_size = 2 bytes
                    | (1u << 20)   // pad_enable
                    | (3u << 22)   // pad_interval: every 16 DWORDs (one 64B row)
                    | (3u << 25);  // pad_amount: 4 DWORDs (16B)
  v8i g1 = {(int)p0,
            (int)((unsigned)V << 16),   // tensor_dim0 = V     (bits 79:48)
            (int)((unsigned)H << 16),   // tensor_dim1 = H     (bits 111:80)
            (int)(32u << 16),           // tile_dim0 = 32      (bits 127:112)
            128,                        // tile_dim1 = 128     (bits 143:128)
            V,                          // tensor_dim0_stride  (bits 207:160)
            0, 0};
  v4i gz = {0, 0, 0, 0};
#if defined(__clang_major__) && __clang_major__ >= 23
  v8i gz8 = {0, 0, 0, 0, 0, 0, 0, 0};
  __builtin_amdgcn_tensor_load_to_lds(g0, g1, gz, gz, gz8, 0);
#else
  __builtin_amdgcn_tensor_load_to_lds(g0, g1, gz, gz, 0);
#endif
#else
  (void)gsrc; (void)ldst;
#endif
}

// ---------------------------------------------------------------------------
// Kernel 0: tiled transpose + f32->bf16 convert:  in[K][N] f32 -> out[N][K] bf16
// ---------------------------------------------------------------------------
__global__ void __launch_bounds__(256)
transpose_cvt_bf16(const float* __restrict__ in, bf16* __restrict__ out,
                   int K, int N) {
  __shared__ float tile[32][33];
  const int tx = threadIdx.x & 31, ty = threadIdx.x >> 5;
  const int k0 = blockIdx.y * 32, n0 = blockIdx.x * 32;
#pragma unroll
  for (int r = 0; r < 4; ++r)
    tile[ty + r * 8][tx] = in[(size_t)(k0 + ty + r * 8) * N + n0 + tx];
  __syncthreads();
#pragma unroll
  for (int r = 0; r < 4; ++r)
    out[(size_t)(n0 + ty + r * 8) * K + k0 + tx] = (bf16)tile[tx][ty + r * 8];
}

// ---------------------------------------------------------------------------
// Kernel 1: zero h ping-pong buffers (incl. pad rows 25..31) + barrier counters
// ---------------------------------------------------------------------------
__global__ void zero_ws(unsigned* __restrict__ hb, int* __restrict__ cnt) {
  const int i = blockIdx.x * 256 + threadIdx.x;
  if (i < 2 * 32 * 1024 / 2) hb[i] = 0u;   // 2 buffers * 32*1024 bf16 = 32768 u32
  if (i < 256) cnt[i] = 0;
}

// ---------------------------------------------------------------------------
// Kernel 2: big GEMM  P = X * W_xh  via bf16 WMMA, double-buffered LDS tiles.
// Block tile 128x128x32, 256 threads (8 waves as 2x4), each wave 64x32 output.
// A-tiles: register-staged (f32->bf16 convert).  B-tiles: TDM async DMA.
// ---------------------------------------------------------------------------
#define BM 128
#define BN 128
#define BK 32
#define LDA 40   // padded LDS row stride (elements): 80B rows -> conflict-free b128

__global__ void __launch_bounds__(256)
gemm_x_wxh(const float* __restrict__ X, const bf16* __restrict__ Wt,
           float* __restrict__ P) {
  __shared__ bf16 As[2][BM * LDA];
  __shared__ bf16 Bs[2][BN * LDA];

  const int tid  = threadIdx.x;
  const int lane = tid & 31;
  const int wave = tid >> 5;
  const int wm   = wave >> 2;          // 0..1  (M sub-block)
  const int wn   = wave & 3;           // 0..3  (N sub-block)
  const int mBase = blockIdx.y * BM;
  const int nBase = blockIdx.x * BN;

  // staging: thread -> (row, k-half)
  const int srow  = tid >> 1;          // 0..127
  const int shalf = (tid & 1) * 16;    // 0 or 16
  const float* ag     = X  + (size_t)(mBase + srow) * V + shalf;
  const bf16*  wtBase = Wt + (size_t)nBase * V;   // row 0 of this N-block
#if !HAVE_TDM
  const bf16*  bg = Wt + (size_t)(nBase + srow) * V + shalf;
#endif

  const int lrow = lane & 15;
  const int lk   = (lane >> 4) * 8;

  const v8f vzero = {0.f, 0.f, 0.f, 0.f, 0.f, 0.f, 0.f, 0.f};
  v8f acc[4][2];
#pragma unroll
  for (int i = 0; i < 4; ++i)
#pragma unroll
    for (int j = 0; j < 2; ++j) acc[i][j] = vzero;

  // ---- stage chunk 0
#if HAVE_TDM
  if (wave == 0) tdm_load_wt_tile(wtBase, &Bs[0][0]);
#endif
  {
    float4 f0 = *(const float4*)(ag + 0);
    float4 f1 = *(const float4*)(ag + 4);
    float4 f2 = *(const float4*)(ag + 8);
    float4 f3 = *(const float4*)(ag + 12);
    v8bf alo = {(bf16)f0.x, (bf16)f0.y, (bf16)f0.z, (bf16)f0.w,
                (bf16)f1.x, (bf16)f1.y, (bf16)f1.z, (bf16)f1.w};
    v8bf ahi = {(bf16)f2.x, (bf16)f2.y, (bf16)f2.z, (bf16)f2.w,
                (bf16)f3.x, (bf16)f3.y, (bf16)f3.z, (bf16)f3.w};
    *(v8bf*)&As[0][srow * LDA + shalf]     = alo;
    *(v8bf*)&As[0][srow * LDA + shalf + 8] = ahi;
#if !HAVE_TDM
    *(v8bf*)&Bs[0][srow * LDA + shalf]     = *(const v8bf*)(bg);
    *(v8bf*)&Bs[0][srow * LDA + shalf + 8] = *(const v8bf*)(bg + 8);
#endif
  }
#if HAVE_TDM
  if (wave == 0) __builtin_amdgcn_s_wait_tensorcnt(0);
#endif
  __syncthreads();

  const int KC = V / BK;  // 1000
#pragma unroll 2
  for (int kc = 0; kc < KC; ++kc) {
    const int buf = kc & 1, nxt = buf ^ 1;
    const bool pre = (kc + 1 < KC);
    float4 f0, f1, f2, f3;
#if !HAVE_TDM
    v8bf b0, b1;
#endif
    if (pre) {  // launch next chunk: TDM for B, register prefetch for A
#if HAVE_TDM
      if (wave == 0)
        tdm_load_wt_tile(wtBase + (size_t)(kc + 1) * BK, &Bs[nxt][0]);
#endif
      const float* a2 = ag + (size_t)(kc + 1) * BK;
      f0 = *(const float4*)(a2 + 0);
      f1 = *(const float4*)(a2 + 4);
      f2 = *(const float4*)(a2 + 8);
      f3 = *(const float4*)(a2 + 12);
#if !HAVE_TDM
      const bf16* b2 = bg + (size_t)(kc + 1) * BK;
      b0 = *(const v8bf*)(b2);
      b1 = *(const v8bf*)(b2 + 8);
#endif
    }
    // compute this chunk from LDS
    v16bf afrag[4], bfrag[2];
#pragma unroll
    for (int mt = 0; mt < 4; ++mt)
      afrag[mt] = make_frag(&As[buf][(wm * 64 + mt * 16 + lrow) * LDA + lk]);
#pragma unroll
    for (int nt = 0; nt < 2; ++nt)
      bfrag[nt] = make_frag(&Bs[buf][(wn * 32 + nt * 16 + lrow) * LDA + lk]);
#pragma unroll
    for (int mt = 0; mt < 4; ++mt)
#pragma unroll
      for (int nt = 0; nt < 2; ++nt)
        acc[mt][nt] = wmma_bf16(afrag[mt], bfrag[nt], acc[mt][nt]);

    if (pre) {  // convert + store prefetched A chunk to the other buffer
      v8bf alo = {(bf16)f0.x, (bf16)f0.y, (bf16)f0.z, (bf16)f0.w,
                  (bf16)f1.x, (bf16)f1.y, (bf16)f1.z, (bf16)f1.w};
      v8bf ahi = {(bf16)f2.x, (bf16)f2.y, (bf16)f2.z, (bf16)f2.w,
                  (bf16)f3.x, (bf16)f3.y, (bf16)f3.z, (bf16)f3.w};
      *(v8bf*)&As[nxt][srow * LDA + shalf]     = alo;
      *(v8bf*)&As[nxt][srow * LDA + shalf + 8] = ahi;
#if !HAVE_TDM
      *(v8bf*)&Bs[nxt][srow * LDA + shalf]     = b0;
      *(v8bf*)&Bs[nxt][srow * LDA + shalf + 8] = b1;
#endif
    }
#if HAVE_TDM
    if (pre && wave == 0) __builtin_amdgcn_s_wait_tensorcnt(0);
#endif
    __syncthreads();
  }

  // epilogue: D layout -> lane<16: (M=r, N=lane); lane>=16: (M=8+r, N=lane-16)
#pragma unroll
  for (int mt = 0; mt < 4; ++mt)
#pragma unroll
    for (int nt = 0; nt < 2; ++nt) {
      const int gn  = nBase + wn * 32 + nt * 16 + lrow;
      const int gm0 = mBase + wm * 64 + mt * 16 + (lane >> 4) * 8;
#pragma unroll
      for (int r = 0; r < 8; ++r)
        P[(size_t)(gm0 + r) * H + gn] = acc[mt][nt][r];
    }
}

// ---------------------------------------------------------------------------
// Kernel 3: recurrence. 32 persistent WGs x 128 threads; each WG owns 32 cols,
// W_hh^T slice (32x1024 bf16 = 64KB) resident in LDS; atomic grid barrier/step.
// ---------------------------------------------------------------------------
DEVINL int wsw(int n, int k) {  // swizzled LDS index (breaks 2KB-stride conflicts)
  return n * 1024 + ((k + ((n & 7) << 3)) & 1023);
}

__global__ void __launch_bounds__(128)
rnn_recurrence(const float* __restrict__ P, const float* __restrict__ bh,
               const bf16* __restrict__ Whht, bf16* __restrict__ hb0,
               bf16* __restrict__ hb1, float* __restrict__ hf,
               int* __restrict__ cnt) {
  __shared__ bf16 Ws[32 * 1024];  // 64KB: this block's 32 columns of W_hh^T
  const int tid   = threadIdx.x;
  const int nBase = blockIdx.x * 32;

  for (int c = tid; c < 32 * 128; c += 128) {
    const int n  = c >> 7;
    const int k8 = (c & 127) * 8;
    *(v8bf*)&Ws[wsw(n, k8)] =
        *(const v8bf*)&Whht[(size_t)(nBase + n) * 1024 + k8];
  }
  __syncthreads();

  const int lane = tid & 31;
  const int wave = tid >> 5;
  const int mt   = wave >> 1;  // 0..1 (M tile: rows 0-15 / 16-31; 25 valid)
  const int ntl  = wave & 1;   // 0..1 (N tile within 32-col slice)
  const int lrow = lane & 15;
  const int lk   = (lane >> 4) * 8;
  const int gn   = nBase + ntl * 16 + lrow;
  const int nloc = ntl * 16 + lrow;
  const float bias = bh[gn];
  const v8f vzero = {0.f, 0.f, 0.f, 0.f, 0.f, 0.f, 0.f, 0.f};

  for (int t = 0; t < TT; ++t) {
    const bf16* hin  = (t & 1) ? hb1 : hb0;
    bf16*       hout = (t & 1) ? hb0 : hb1;
    v8f acc = vzero;
#pragma unroll 4
    for (int kc = 0; kc < 32; ++kc) {
      const int kk = kc * 32;
      v16bf a = make_frag(hin + (mt * 16 + lrow) * 1024 + kk + lk);
      v8bf blo = *(const v8bf*)&Ws[wsw(nloc, kk + lk)];
      v8bf bhi = *(const v8bf*)&Ws[wsw(nloc, kk + lk + 16)];
      v16bf b = __builtin_shufflevector(blo, bhi,
                  0,1,2,3,4,5,6,7,8,9,10,11,12,13,14,15);
      acc = wmma_bf16(a, b, acc);
    }
    const int m0 = mt * 16 + (lane >> 4) * 8;
#pragma unroll
    for (int r = 0; r < 8; ++r) {
      const int m = m0 + r;
      if (m < BSZ) {
        float v = tanhf(acc[r] + P[(size_t)(t * BSZ + m) * H + gn] + bias);
        hout[m * 1024 + gn] = (bf16)v;
        if (t == TT - 1) hf[m * 1024 + gn] = v;
      }
    }
    // ---- grid-wide barrier (release writes, acquire for next step's reads)
    __syncthreads();
    if (tid == 0) {
      __threadfence();
      atomicAdd(&cnt[t], 1);
      while (__hip_atomic_load(&cnt[t], __ATOMIC_ACQUIRE,
                               __HIP_MEMORY_SCOPE_AGENT) < (int)gridDim.x)
        __builtin_amdgcn_s_sleep(1);
    }
    __syncthreads();
    __threadfence();  // invalidate per-CU caches in every wave
  }
}

// ---------------------------------------------------------------------------
// Kernel 4: mean-pool over batch, FC(1024->2), softmax
// ---------------------------------------------------------------------------
__global__ void __launch_bounds__(256)
fc_softmax(const float* __restrict__ hf, const float* __restrict__ fw,
           const float* __restrict__ fb, float* __restrict__ out) {
  __shared__ float r0[256], r1[256];
  float s0 = 0.f, s1 = 0.f;
  for (int n = threadIdx.x; n < H; n += 256) {
    float pooled = 0.f;
#pragma unroll
    for (int b = 0; b < BSZ; ++b) pooled += hf[b * 1024 + n];
    pooled *= (1.0f / (float)BSZ);
    s0 += pooled * fw[n * 2 + 0];
    s1 += pooled * fw[n * 2 + 1];
  }
  r0[threadIdx.x] = s0;
  r1[threadIdx.x] = s1;
  __syncthreads();
  for (int s = 128; s > 0; s >>= 1) {
    if ((int)threadIdx.x < s) {
      r0[threadIdx.x] += r0[threadIdx.x + s];
      r1[threadIdx.x] += r1[threadIdx.x + s];
    }
    __syncthreads();
  }
  if (threadIdx.x == 0) {
    float l0 = r0[0] + fb[0], l1 = r1[0] + fb[1];
    float mx = fmaxf(l0, l1);
    float e0 = __expf(l0 - mx), e1 = __expf(l1 - mx);
    float inv = 1.f / (e0 + e1);
    out[0] = e0 * inv;
    out[1] = e1 * inv;
  }
}

// ---------------------------------------------------------------------------
// Launch
// ---------------------------------------------------------------------------
extern "C" void kernel_launch(void* const* d_in, const int* in_sizes, int n_in,
                              void* d_out, int out_size, void* d_ws,
                              size_t ws_size, hipStream_t stream) {
  (void)in_sizes; (void)n_in; (void)out_size; (void)ws_size;
  const float* X   = (const float*)d_in[0];
  const float* Wxh = (const float*)d_in[1];
  const float* Whh = (const float*)d_in[2];
  const float* bh  = (const float*)d_in[3];
  const float* fw  = (const float*)d_in[4];
  const float* fb  = (const float*)d_in[5];
  float* out = (float*)d_out;

  // workspace layout (all 256B-aligned)
  char* ws = (char*)d_ws;
  constexpr size_t OFF_P    = 0;                                   // 26,214,400
  constexpr size_t OFF_WT   = OFF_P    + (size_t)M * H * 4;        // 65,536,000
  constexpr size_t OFF_WHHT = OFF_WT   + (size_t)H * V * 2;        //  2,097,152
  constexpr size_t OFF_HB0  = OFF_WHHT + (size_t)H * H * 2;        //     65,536
  constexpr size_t OFF_HB1  = OFF_HB0  + (size_t)32 * 1024 * 2;    //     65,536
  constexpr size_t OFF_HF   = OFF_HB1  + (size_t)32 * 1024 * 2;    //    131,072
  constexpr size_t OFF_CNT  = OFF_HF   + (size_t)32 * 1024 * 4;    //      1,024
  float* P    = (float*)(ws + OFF_P);
  bf16*  Wt   = (bf16*)(ws + OFF_WT);
  bf16*  Whht = (bf16*)(ws + OFF_WHHT);
  bf16*  hb0  = (bf16*)(ws + OFF_HB0);
  bf16*  hb1  = (bf16*)(ws + OFF_HB1);
  float* hf   = (float*)(ws + OFF_HF);
  int*   cnt  = (int*)(ws + OFF_CNT);

  transpose_cvt_bf16<<<dim3(H / 32, V / 32), 256, 0, stream>>>(Wxh, Wt, V, H);
  transpose_cvt_bf16<<<dim3(H / 32, H / 32), 256, 0, stream>>>(Whh, Whht, H, H);
  zero_ws<<<128, 256, 0, stream>>>((unsigned*)hb0, cnt);
  gemm_x_wxh<<<dim3(H / BN, M / BM), 256, 0, stream>>>(X, Wt, P);
  rnn_recurrence<<<32, 128, 0, stream>>>(P, bh, Whht, hb0, hb1, hf, cnt);
  fc_softmax<<<1, 256, 0, stream>>>(hf, fw, fb, out);
}
